// SpatialSelfAttention_3942779977829
// MI455X (gfx1250) — matrix operand order repaired and checked by reading, not compiled, not run on hardware
//
#include <hip/hip_runtime.h>
#include <hip/hip_bf16.h>

// ---------------- types ----------------
typedef __bf16 bf16;
typedef __attribute__((ext_vector_type(16))) __bf16 v16bf;
typedef __attribute__((ext_vector_type(8)))  __bf16 v8bf;
typedef __attribute__((ext_vector_type(8)))  float  v8f;

#define BATCH 4
#define CCH   512
#define NPIX  4096   // 64*64

__device__ inline v8f wmma_bf16(v16bf a, v16bf b, v8f c) {
  return __builtin_amdgcn_wmma_f32_16x16x32_bf16(
      /*neg_a=*/false, a, /*neg_b=*/false, b,
      /*c_mod=*/(short)0, c, /*reuse_a=*/false, /*reuse_b=*/false);
}

// A-fragment (16x32 bf16, M x K), source row-major (m,k), k contiguous, stride ld.
// ISA layout: lane<16 -> M=lane, elems 0..7 = K 0..7, elems 8..15 = K 16..23
//             lane>=16 -> M=lane-16, elems = K 8..15 and K 24..31
__device__ inline v16bf load_a_frag(const bf16* base, int ld, int k0) {
  const int lane = threadIdx.x & 31;
  const int row  = lane & 15;
  const int kb   = k0 + ((lane >> 4) << 3);
  const bf16* p = base + (size_t)row * ld + kb;
  union { v16bf v; v8bf h[2]; } u;
  u.h[0] = *(const v8bf*)(p);
  u.h[1] = *(const v8bf*)(p + 16);
  return u.v;
}

// B-fragment (32x16 bf16, K x N), source (n,k) with k contiguous, stride ld.
// ISA layout: lane<16 -> N=lane, elems = K 0..7, K 16..23 ; lane>=16 -> N=lane-16, K 8..15, K 24..31
__device__ inline v16bf load_b_frag(const bf16* base, int ld, int k0) {
  const int lane = threadIdx.x & 31;
  const int col  = lane & 15;
  const int kb   = k0 + ((lane >> 4) << 3);
  const bf16* p = base + (size_t)col * ld + kb;
  union { v16bf v; v8bf h[2]; } u;
  u.h[0] = *(const v8bf*)(p);
  u.h[1] = *(const v8bf*)(p + 16);
  return u.v;
}

// ---------------- kernel: weight conversion (fp32 -> bf16) ----------------
// Wbf = [wq*s ; wk ; wv]  (1536 x 512 row-major), Wobf = wo (512 x 512)
__global__ __launch_bounds__(256) void ssa_convert_w(
    const float* __restrict__ wq, const float* __restrict__ wk,
    const float* __restrict__ wv, const float* __restrict__ wo,
    bf16* __restrict__ Wbf, bf16* __restrict__ Wobf) {
  const int idx = blockIdx.x * 256 + threadIdx.x;
  const float s = 0.044194173824159216f;  // 1/sqrt(512)
  const int WSZ = CCH * CCH;
  if (idx < 3 * WSZ) {
    float v;
    if (idx < WSZ)          v = wq[idx] * s;
    else if (idx < 2 * WSZ) v = wk[idx - WSZ];
    else                    v = wv[idx - 2 * WSZ];
    Wbf[idx] = (bf16)v;
  } else if (idx < 4 * WSZ) {
    const int k = idx - 3 * WSZ;
    Wobf[k] = (bf16)wo[k];
  }
}

// ---------------- kernel: x (b,c,n) f32 -> Xt (b,n,c) bf16 ----------------
__global__ __launch_bounds__(256) void ssa_transpose_x(
    const float* __restrict__ x, bf16* __restrict__ Xt) {
  __shared__ float tile[32][33];
  const int b  = blockIdx.z;
  const int c0 = blockIdx.y * 32;
  const int n0 = blockIdx.x * 32;
  const int tr = threadIdx.x >> 5;   // 0..7
  const int tc = threadIdx.x & 31;
  const float* src = x + ((size_t)b * CCH + c0) * NPIX + n0;
  for (int r = tr; r < 32; r += 8)
    tile[r][tc] = src[(size_t)r * NPIX + tc];
  __syncthreads();
  bf16* dst = Xt + ((size_t)b * NPIX + n0) * CCH + c0;
  for (int r = tr; r < 32; r += 8)
    dst[(size_t)r * CCH + tc] = (bf16)tile[tc][r];
}

// ---------------- kernel: QKV projection GEMM ----------------
// D(m,n) = sum_c Wbf(m,c) * Xt(n,c) ; m in [0,1536), n in [0,4096) per batch
// Block tile 64(m) x 256(n); 8 waves as 2(m) x 4(n); wave tile 32 x 64.
__global__ __launch_bounds__(256) void ssa_qkv_gemm(
    const bf16* __restrict__ Wbf, const bf16* __restrict__ Xt,
    const float* __restrict__ bq, const float* __restrict__ bk,
    const float* __restrict__ bv,
    bf16* __restrict__ Qt, bf16* __restrict__ Kt, bf16* __restrict__ Vm) {
  const int b    = blockIdx.z;
  const int wave = threadIdx.x >> 5;
  const int lane = threadIdx.x & 31;
  const int m0 = blockIdx.x * 64 + (wave >> 2) * 32;
  const int n0 = blockIdx.y * 256 + (wave & 3) * 64;
  const float s = 0.044194173824159216f;  // 1/sqrt(512)

  const bf16* A  = Wbf + (size_t)m0 * CCH;                  // (m,c)
  const bf16* Bm = Xt + ((size_t)b * NPIX + n0) * CCH;      // (n,c)

  v8f acc[2][4] = {};
  for (int k0 = 0; k0 < CCH; k0 += 32) {
    v16bf a0 = load_a_frag(A, CCH, k0);
    v16bf a1 = load_a_frag(A + (size_t)16 * CCH, CCH, k0);
    v16bf bb[4];
#pragma unroll
    for (int j = 0; j < 4; j++)
      bb[j] = load_b_frag(Bm + (size_t)j * 16 * CCH, CCH, k0);
#pragma unroll
    for (int j = 0; j < 4; j++) {
      acc[0][j] = wmma_bf16(a0, bb[j], acc[0][j]);
      acc[1][j] = wmma_bf16(a1, bb[j], acc[1][j]);
    }
  }

  const int col = lane & 15;
  const int rb  = (lane >> 4) * 8;
#pragma unroll
  for (int i = 0; i < 2; i++) {
    const int mt = m0 + i * 16;   // 16-row tile; entirely inside one of Q/K/V
#pragma unroll
    for (int j = 0; j < 4; j++) {
      const int nt = n0 + j * 16 + col;
      if (mt < CCH) {                      // Q -> Qt (b,n,c), scale folded
        v8bf o;
#pragma unroll
        for (int r = 0; r < 8; r++)
          o[r] = (bf16)(acc[i][j][r] + bq[mt + rb + r] * s);
        *(v8bf*)(Qt + ((size_t)b * NPIX + nt) * CCH + mt + rb) = o;
      } else if (mt < 2 * CCH) {           // K -> Kt (b,n,c)
        const int m = mt - CCH;
        v8bf o;
#pragma unroll
        for (int r = 0; r < 8; r++)
          o[r] = (bf16)(acc[i][j][r] + bk[m + rb + r]);
        *(v8bf*)(Kt + ((size_t)b * NPIX + nt) * CCH + m + rb) = o;
      } else {                             // V -> Vm (b,c,n) natural
        const int m = mt - 2 * CCH;
#pragma unroll
        for (int r = 0; r < 8; r++)
          Vm[((size_t)b * CCH + m + rb + r) * NPIX + nt] =
              (bf16)(acc[i][j][r] + bv[m + rb + r]);
      }
    }
  }
}

// ---------------- kernel: flash attention ----------------
// One block = 16 query rows. 8 waves. Per key-chunk of 64:
//  step1: S(16x64) = Q_i * K_j^T   (wave = jtile(4) x khalf(2), LDS reduce)
//  step2: online softmax (shfl reductions within half-waves)
//  step3: O^T(c,i) += V(c,j) * P^T(j,i)  (waves split C, 64 channels each)
__global__ __launch_bounds__(256) void ssa_attention(
    const bf16* __restrict__ Qt, const bf16* __restrict__ Kt,
    const bf16* __restrict__ Vm, bf16* __restrict__ Ot) {
  const int b    = blockIdx.y;
  const int i0   = blockIdx.x * 16;
  const int tid  = threadIdx.x;
  const int wave = tid >> 5;
  const int lane = tid & 31;

  __shared__ alignas(16) float Spart[4][16][16];
  __shared__ alignas(16) float Sfull[16][64];
  __shared__ alignas(16) bf16  Pbf[16][64];
  __shared__ float m_run[16], l_run[16], alpha_sh[16];

  if (tid < 16) { m_run[tid] = -3.0e38f; l_run[tid] = 0.0f; }

  const int jt = wave & 3;      // j sub-tile for step1
  const int kh = wave >> 2;     // k half for step1
  const int c0 = wave * 64;     // channel range for step3

  const bf16* Qbase = Qt + ((size_t)b * NPIX + i0) * CCH;   // (i,c)
  // cache this wave's Q fragments for its k-half (16 x 256)
  v16bf qfrag[8];
#pragma unroll
  for (int kk = 0; kk < 8; kk++)
    qfrag[kk] = load_a_frag(Qbase, CCH, kh * 256 + kk * 32);

  v8f acc[4] = {};

  for (int j0 = 0; j0 < NPIX; j0 += 64) {
    // ---- step 1: partial S for this wave's (jt, kh) ----
    v8f sv = {};
    const bf16* Kbase = Kt + ((size_t)b * NPIX + j0 + jt * 16) * CCH;
#pragma unroll
    for (int kk = 0; kk < 8; kk++) {
      v16bf bb = load_b_frag(Kbase, CCH, kh * 256 + kk * 32);
      sv = wmma_bf16(qfrag[kk], bb, sv);
    }
    __syncthreads();   // also: previous step3 done reading Pbf
    {
      const int col = lane & 15, rbb = (lane >> 4) * 8;
      if (kh == 1) {
#pragma unroll
        for (int r = 0; r < 8; r++) Spart[jt][rbb + r][col] = sv[r];
      }
    }
    __syncthreads();
    {
      const int col = lane & 15, rbb = (lane >> 4) * 8;
      if (kh == 0) {
#pragma unroll
        for (int r = 0; r < 8; r++)
          Sfull[rbb + r][jt * 16 + col] = sv[r] + Spart[jt][rbb + r][col];
      }
    }
    __syncthreads();
    // ---- step 2: online softmax over Sfull rows ----
    {
      const int r = tid >> 4, q = tid & 15;   // 16 threads per row
      float v0 = Sfull[r][q * 4 + 0], v1 = Sfull[r][q * 4 + 1];
      float v2 = Sfull[r][q * 4 + 2], v3 = Sfull[r][q * 4 + 3];
      float mx = fmaxf(fmaxf(v0, v1), fmaxf(v2, v3));
      for (int off = 8; off >= 1; off >>= 1)
        mx = fmaxf(mx, __shfl_xor(mx, off, 16));
      const float m_old = m_run[r];
      const float m_new = fmaxf(m_old, mx);
      const float p0 = __expf(v0 - m_new), p1 = __expf(v1 - m_new);
      const float p2 = __expf(v2 - m_new), p3 = __expf(v3 - m_new);
      float sm = p0 + p1 + p2 + p3;
      for (int off = 8; off >= 1; off >>= 1)
        sm += __shfl_xor(sm, off, 16);
      Pbf[r][q * 4 + 0] = (bf16)p0; Pbf[r][q * 4 + 1] = (bf16)p1;
      Pbf[r][q * 4 + 2] = (bf16)p2; Pbf[r][q * 4 + 3] = (bf16)p3;
      if (q == 0) {
        const float alpha = __expf(m_old - m_new);
        l_run[r] = l_run[r] * alpha + sm;
        m_run[r] = m_new;
        alpha_sh[r] = alpha;
      }
    }
    __syncthreads();
    // ---- step 3: rescale + accumulate O^T ----
    {
      const float alpha = alpha_sh[lane & 15];  // lane's fixed i-column
#pragma unroll
      for (int t = 0; t < 4; t++)
#pragma unroll
        for (int r = 0; r < 8; r++) acc[t][r] *= alpha;
    }
    const bf16* Vbase = Vm + ((size_t)b * CCH + c0) * NPIX + j0;  // (c,j)
#pragma unroll
    for (int ks = 0; ks < 2; ks++) {
      const int kj = ks * 32;
      v16bf pb = load_b_frag(&Pbf[0][0], 64, kj);  // P^T : n=i, k=j
#pragma unroll
      for (int t = 0; t < 4; t++) {
        v16bf av = load_a_frag(Vbase + (size_t)t * 16 * NPIX, NPIX, kj);
        acc[t] = wmma_bf16(av, pb, acc[t]);
      }
    }
  }
  __syncthreads();
  // ---- finalize: O /= l ; store Ot (b,n,c) bf16 ----
  const float linv = 1.0f / l_run[lane & 15];
  const int i  = i0 + (lane & 15);
  const int rb = (lane >> 4) * 8;
#pragma unroll
  for (int t = 0; t < 4; t++) {
    v8bf o;
#pragma unroll
    for (int r = 0; r < 8; r++) o[r] = (bf16)(acc[t][r] * linv);
    *(v8bf*)(Ot + ((size_t)b * NPIX + i) * CCH + c0 + t * 16 + rb) = o;
  }
}

// ---------------- kernel: output projection + residual ----------------
// out(b,o,n) = x(b,o,n) + bo[o] + sum_c Wobf(o,c) * Ot(b,n,c)
__global__ __launch_bounds__(256) void ssa_out_gemm(
    const bf16* __restrict__ Wobf, const bf16* __restrict__ Ot,
    const float* __restrict__ bo, const float* __restrict__ x,
    float* __restrict__ out) {
  const int b    = blockIdx.z;
  const int wave = threadIdx.x >> 5;
  const int lane = threadIdx.x & 31;
  const int m0 = blockIdx.x * 64 + (wave >> 2) * 32;
  const int n0 = blockIdx.y * 256 + (wave & 3) * 64;

  const bf16* A  = Wobf + (size_t)m0 * CCH;
  const bf16* Bm = Ot + ((size_t)b * NPIX + n0) * CCH;

  v8f acc[2][4] = {};
  for (int k0 = 0; k0 < CCH; k0 += 32) {
    v16bf a0 = load_a_frag(A, CCH, k0);
    v16bf a1 = load_a_frag(A + (size_t)16 * CCH, CCH, k0);
    v16bf bb[4];
#pragma unroll
    for (int j = 0; j < 4; j++)
      bb[j] = load_b_frag(Bm + (size_t)j * 16 * CCH, CCH, k0);
#pragma unroll
    for (int j = 0; j < 4; j++) {
      acc[0][j] = wmma_bf16(a0, bb[j], acc[0][j]);
      acc[1][j] = wmma_bf16(a1, bb[j], acc[1][j]);
    }
  }

  const int col = lane & 15;
  const int rb  = (lane >> 4) * 8;
#pragma unroll
  for (int i = 0; i < 2; i++) {
    const int mt = m0 + i * 16;
#pragma unroll
    for (int j = 0; j < 4; j++) {
      const int nt = n0 + j * 16 + col;
#pragma unroll
      for (int r = 0; r < 8; r++) {
        const int o = mt + rb + r;
        const size_t idx = ((size_t)b * CCH + o) * NPIX + nt;
        out[idx] = acc[i][j][r] + bo[o] + x[idx];
      }
    }
  }
}

// ---------------- launch ----------------
extern "C" void kernel_launch(void* const* d_in, const int* in_sizes, int n_in,
                              void* d_out, int out_size, void* d_ws, size_t ws_size,
                              hipStream_t stream) {
  const float* x  = (const float*)d_in[0];
  const float* wq = (const float*)d_in[1];
  const float* bq = (const float*)d_in[2];
  const float* wk = (const float*)d_in[3];
  const float* bk = (const float*)d_in[4];
  const float* wv = (const float*)d_in[5];
  const float* bv = (const float*)d_in[6];
  const float* wo = (const float*)d_in[7];
  const float* bo = (const float*)d_in[8];
  float* out = (float*)d_out;

  // workspace carve-up (all bf16, all sizes 16B-aligned)
  const size_t TEN = (size_t)BATCH * NPIX * CCH;  // 8M elements
  char* ws = (char*)d_ws;
  bf16* Xt   = (bf16*)ws;  ws += TEN * sizeof(bf16);
  bf16* Qt   = (bf16*)ws;  ws += TEN * sizeof(bf16);
  bf16* Kt   = (bf16*)ws;  ws += TEN * sizeof(bf16);
  bf16* Vm   = (bf16*)ws;  ws += TEN * sizeof(bf16);
  bf16* Ot   = (bf16*)ws;  ws += TEN * sizeof(bf16);
  bf16* Wbf  = (bf16*)ws;  ws += (size_t)3 * CCH * CCH * sizeof(bf16);
  bf16* Wobf = (bf16*)ws;  ws += (size_t)CCH * CCH * sizeof(bf16);

  ssa_convert_w<<<dim3(4 * CCH * CCH / 256), 256, 0, stream>>>(
      wq, wk, wv, wo, Wbf, Wobf);
  ssa_transpose_x<<<dim3(NPIX / 32, CCH / 32, BATCH), 256, 0, stream>>>(x, Xt);
  ssa_qkv_gemm<<<dim3(3 * CCH / 64, NPIX / 256, BATCH), 256, 0, stream>>>(
      Wbf, Xt, bq, bk, bv, Qt, Kt, Vm);
  ssa_attention<<<dim3(NPIX / 16, BATCH), 256, 0, stream>>>(Qt, Kt, Vm, Ot);
  ssa_out_gemm<<<dim3(CCH / 64, NPIX / 256, BATCH), 256, 0, stream>>>(
      Wobf, Ot, bo, x, out);
}
